// KrauseQwenAttention_49143015801161
// MI455X (gfx1250) — matrix operand order repaired and checked by reading, not compiled, not run on hardware
//
#include <hip/hip_runtime.h>
#include <hip/hip_bf16.h>

typedef __attribute__((ext_vector_type(16))) _Float16 v16h;
typedef __attribute__((ext_vector_type(8)))  _Float16 v8h;
typedef __attribute__((ext_vector_type(8)))  float    v8f;

constexpr int kT   = 2048;
constexpr int kHID = 1024;
constexpr int kH   = 16;
constexpr int kHKV = 8;
constexpr int kD   = 64;
constexpr int kWIN = 128;
constexpr int kTOPK = 32;
constexpr float kSCALE = 0.125f;               // 64^-0.5
constexpr float kNEG = -3.402823466e38f;       // finfo(f32).min

// ---------------- helpers ----------------

__device__ inline v16h combine16(v8h lo, v8h hi2) {
  v16h r;
#pragma unroll
  for (int i = 0; i < 8; ++i) { r[i] = lo[i]; r[i + 8] = hi2[i]; }
  return r;
}

// A (16x32 f16): lane L -> row L&15; halves: K = 16*hi + 0..15
// -> two contiguous 16B chunks per lane (global_load_b128 x2)
__device__ inline v16h load_a_frag16(const _Float16* __restrict__ A, int lda,
                                     int m0, int k0, int lane) {
  int row = m0 + (lane & 15);
  int hi  = (lane >> 4) & 1;
  const _Float16* p = A + (size_t)row * lda + k0 + 16 * hi;
  v8h lo  = *(const v8h*)p;
  v8h hp  = *(const v8h*)(p + 8);
  return combine16(lo, hp);
}

// B (32x16 f16) with B[k][n] = W[n][k] (X @ W^T): lane -> col n, K = 16*hi + i
// -> two contiguous 16B chunks per lane
__device__ inline v16h load_b_frag_wT16(const _Float16* __restrict__ W, int ldw,
                                        int k0, int n0, int lane) {
  int n  = n0 + (lane & 15);
  int hi = (lane >> 4) & 1;
  const _Float16* p = W + (size_t)n * ldw + k0 + 16 * hi;
  v8h lo = *(const v8h*)p;
  v8h hp = *(const v8h*)(p + 8);
  return combine16(lo, hp);
}

// B (32x16 f16) from row-major K x N f16 matrix (P @ V): strided, scalar f16 loads
__device__ inline v16h load_b_frag_rm16(const _Float16* __restrict__ Bm, int ldb,
                                        int k0, int n0, int lane) {
  int n  = n0 + (lane & 15);
  int hi = (lane >> 4) & 1;
  v16h b;
#pragma unroll
  for (int i = 0; i < 16; ++i)
    b[i] = Bm[(size_t)(k0 + 16 * hi + i) * ldb + n];
  return b;
}

__device__ inline float wmax16(float x) {
#pragma unroll
  for (int o = 1; o < 16; o <<= 1) x = fmaxf(x, __shfl_xor(x, o, 32));
  return x;
}
__device__ inline float wsum16(float x) {
#pragma unroll
  for (int o = 1; o < 16; o <<= 1) x += __shfl_xor(x, o, 32);
  return x;
}
__device__ inline float wmax32(float x) {
#pragma unroll
  for (int o = 1; o < 32; o <<= 1) x = fmaxf(x, __shfl_xor(x, o, 32));
  return x;
}
__device__ inline float wsum32(float x) {
#pragma unroll
  for (int o = 1; o < 32; o <<= 1) x += __shfl_xor(x, o, 32);
  return x;
}

// ---------------- f32 -> f16 convert ----------------
__global__ void f32_to_f16(const float* __restrict__ src, _Float16* __restrict__ dst, int n) {
  int i = blockIdx.x * blockDim.x + threadIdx.x;
  int i4 = i * 4;
  if (i4 + 3 < n) {
    float4 v = *(const float4*)(src + i4);
    dst[i4 + 0] = (_Float16)v.x;
    dst[i4 + 1] = (_Float16)v.y;
    dst[i4 + 2] = (_Float16)v.z;
    dst[i4 + 3] = (_Float16)v.w;
  } else {
    for (int j = i4; j < n; ++j) dst[j] = (_Float16)src[j];
  }
}

// ---------------- GEMM: Y[M x N] = X[M x K] @ W[N x K]^T + bias ----------------
// 16x64 tile per wave: one A-frag reused across 4 WMMAs per k-step.
__global__ __launch_bounds__(32)
void gemm_xwT16(const _Float16* __restrict__ X, const _Float16* __restrict__ W,
                const float* __restrict__ bias, float* __restrict__ Y, int N, int K) {
  int m0 = blockIdx.x * 16;
  int n0 = blockIdx.y * 64;
  int lane = threadIdx.x;
  v8f acc[4] = {};
  for (int k0 = 0; k0 < K; k0 += 32) {
    if (k0 + 32 < K)
      __builtin_prefetch(&X[(size_t)(m0 + (lane & 15)) * K + k0 + 32], 0, 1);
    v16h a = load_a_frag16(X, K, m0, k0, lane);
#pragma unroll
    for (int nb = 0; nb < 4; ++nb) {
      v16h b = load_b_frag_wT16(W, K, k0, n0 + nb * 16, lane);
      acc[nb] = __builtin_amdgcn_wmma_f32_16x16x32_f16(false, a, false, b, (short)0, acc[nb], false, false);
    }
  }
  int hi = lane >> 4;
#pragma unroll
  for (int nb = 0; nb < 4; ++nb) {
    int col = n0 + nb * 16 + (lane & 15);
    float bv = bias ? bias[col] : 0.0f;
#pragma unroll
    for (int r = 0; r < 8; ++r)
      Y[(size_t)(m0 + r + 8 * hi) * N + col] = acc[nb][r] + bv;
  }
}

// ---------------- RoPE (in place on [T, nheads*64] f32 buffer) ----------------
__global__ void rope_kernel(float* __restrict__ X, const float* __restrict__ c,
                            const float* __restrict__ s, int nheads) {
  int idx = blockIdx.x * blockDim.x + threadIdx.x;
  int total = kT * nheads * 32;
  if (idx >= total) return;
  int d = idx & 31;
  int h = (idx >> 5) % nheads;
  int t = idx / (32 * nheads);
  float c1 = c[t * kD + d],      s1 = s[t * kD + d];
  float c2 = c[t * kD + d + 32], s2 = s[t * kD + d + 32];
  float* row = X + (size_t)t * nheads * kD + h * kD;
  float x1 = row[d], x2 = row[d + 32];
  row[d]      = x1 * c1 - x2 * s1;
  row[d + 32] = x2 * c2 + x1 * s2;
}

// ---------------- attn_v pass 1: per-row (max, sumexp), flash-style ----------------
__global__ __launch_bounds__(32)
void attnv_stats(const _Float16* __restrict__ Q, const _Float16* __restrict__ K,
                 float* __restrict__ rowm, float* __restrict__ rowl) {
  int h = blockIdx.y, qt = blockIdx.x, lane = threadIdx.x;
  int m0 = qt * 16, hi = lane >> 4;
  const _Float16* Qh = Q + h * kD;
  const _Float16* Kh = K + (h >> 1) * kD;
  float m[8], l[8];
#pragma unroll
  for (int r = 0; r < 8; ++r) { m[r] = kNEG; l[r] = 0.0f; }
  for (int kt = 0; kt <= qt; ++kt) {
    int k0c = kt * 16;
    v8f sc = {};
#pragma unroll
    for (int kk0 = 0; kk0 < kD; kk0 += 32) {
      v16h a = load_a_frag16(Qh, kH * kD, m0, kk0, lane);
      v16h b = load_b_frag_wT16(Kh, kHKV * kD, kk0, k0c, lane);
      sc = __builtin_amdgcn_wmma_f32_16x16x32_f16(false, a, false, b, (short)0, sc, false, false);
    }
    int col = k0c + (lane & 15);
#pragma unroll
    for (int r = 0; r < 8; ++r) {
      int row = m0 + r + 8 * hi;
      float sv = (col <= row) ? sc[r] * kSCALE : kNEG;
      float tmax = wmax16(sv);
      float nm = fmaxf(m[r], tmax);
      float ts = wsum16(__expf(sv - nm));
      l[r] = l[r] * __expf(m[r] - nm) + ts;
      m[r] = nm;
    }
  }
  if ((lane & 15) == 0) {
#pragma unroll
    for (int r = 0; r < 8; ++r) {
      int row = m0 + r + 8 * hi;
      rowm[h * kT + row] = m[r];
      rowl[h * kT + row] = l[r];
    }
  }
}

// ---------------- attn_v pass 2: recompute, normalize, write attn, P@V ----------------
__global__ __launch_bounds__(32)
void attnv_out(const _Float16* __restrict__ Q, const _Float16* __restrict__ K,
               const _Float16* __restrict__ V, const float* __restrict__ rowm,
               const float* __restrict__ rowl, float* __restrict__ attn,
               float* __restrict__ outv) {
  __shared__ __align__(16) _Float16 pbuf[16 * 32];
  int h = blockIdx.y, qt = blockIdx.x, lane = threadIdx.x;
  int m0 = qt * 16, hi = lane >> 4;
  const _Float16* Qh = Q + h * kD;
  const _Float16* Kh = K + (h >> 1) * kD;
  const _Float16* Vh = V + (h >> 1) * kD;
  float* attn_h = attn + (size_t)h * kT * kT;
  float mrow[8], linv[8];
#pragma unroll
  for (int r = 0; r < 8; ++r) {
    int row = m0 + r + 8 * hi;
    mrow[r] = rowm[h * kT + row];
    linv[r] = 1.0f / rowl[h * kT + row];
  }
  v8f acc[4] = {};
  for (int kt = 0; kt < kT / 16; kt += 2) {
    bool any = (kt * 16 <= m0 + 15);   // uniform across wave
#pragma unroll
    for (int sub = 0; sub < 2; ++sub) {
      int k0c = (kt + sub) * 16;
      int col = k0c + (lane & 15);
      bool act = (k0c <= m0 + 15);     // uniform
      float pvals[8];
      if (act) {
        v8f sc = {};
#pragma unroll
        for (int kk0 = 0; kk0 < kD; kk0 += 32) {
          v16h a = load_a_frag16(Qh, kH * kD, m0, kk0, lane);
          v16h b = load_b_frag_wT16(Kh, kHKV * kD, kk0, k0c, lane);
          sc = __builtin_amdgcn_wmma_f32_16x16x32_f16(false, a, false, b, (short)0, sc, false, false);
        }
#pragma unroll
        for (int r = 0; r < 8; ++r) {
          int row = m0 + r + 8 * hi;
          pvals[r] = (col <= row) ? __expf(sc[r] * kSCALE - mrow[r]) * linv[r] : 0.0f;
        }
      } else {
#pragma unroll
        for (int r = 0; r < 8; ++r) pvals[r] = 0.0f;
      }
#pragma unroll
      for (int r = 0; r < 8; ++r) {
        int row = m0 + r + 8 * hi;
        attn_h[(size_t)row * kT + col] = pvals[r];
        if (any) pbuf[(r + 8 * hi) * 32 + sub * 16 + (lane & 15)] = (_Float16)pvals[r];
      }
    }
    if (any) {
      __syncthreads();
      // rebuild A-frag from row-major P tile: two contiguous 16B LDS chunks per lane
      const v8h* pb = (const v8h*)pbuf;
      int arow = lane & 15;
      v8h lo = pb[arow * 4 + 2 * hi];        // halves [16*hi .. 16*hi+7]
      v8h hp = pb[arow * 4 + 2 * hi + 1];    // halves [16*hi+8 .. 16*hi+15]
      v16h a = combine16(lo, hp);
#pragma unroll
      for (int nb = 0; nb < 4; ++nb) {
        v16h b = load_b_frag_rm16(Vh, kHKV * kD, kt * 16, nb * 16, lane);
        acc[nb] = __builtin_amdgcn_wmma_f32_16x16x32_f16(false, a, false, b, (short)0, acc[nb], false, false);
      }
      __syncthreads();
    }
  }
#pragma unroll
  for (int nb = 0; nb < 4; ++nb)
#pragma unroll
    for (int r = 0; r < 8; ++r)
      outv[(size_t)(m0 + r + 8 * hi) * kHID + h * kD + nb * 16 + (lane & 15)] = acc[nb][r];
}

// ---------------- k-branch: windowed (<=128 keys) dist attention + top-32 ----------------
__global__ __launch_bounds__(32)
void attnk_kernel(const float* __restrict__ QK, const float* __restrict__ KK,
                  const float* __restrict__ VK, const float* __restrict__ log_sigma,
                  float* __restrict__ outk) {
  __shared__ float qrow[kD];
  __shared__ float pshare[kWIN];
  int h = blockIdx.y, q = blockIdx.x, lane = threadIdx.x;
  const float* Qh = QK + (size_t)q * kHID + h * kD;
  const float* Kh = KK + (h >> 1) * kD;
  const float* Vh = VK + (h >> 1) * kD;
  qrow[lane] = Qh[lane];
  qrow[lane + 32] = Qh[lane + 32];
  __syncthreads();
  float sigma = __expf(log_sigma[h]);
  float inv = -1.0f / (2.0f * sigma * sigma);
  int kstart = (q >= kWIN - 1) ? q - (kWIN - 1) : 0;
  float sc[4];
#pragma unroll
  for (int i = 0; i < 4; ++i) {
    int key = kstart + lane + 32 * i;
    float s = kNEG;
    if (key <= q) {
      const float* krow = Kh + (size_t)key * (kHKV * kD);
      float dot = 0.0f, ksq = 0.0f, qsq = 0.0f;
      for (int d = 0; d < kD; ++d) {
        float kd = krow[d], qd = qrow[d];
        dot += qd * kd; ksq += kd * kd; qsq += qd * qd;
      }
      s = (qsq + ksq - 2.0f * dot) * inv;
    }
    sc[i] = s;
  }
  // 32nd-largest via iterative max extraction
  float rem[4];
#pragma unroll
  for (int i = 0; i < 4; ++i) rem[i] = sc[i];
  float thresh = kNEG;
  for (int it = 0; it < kTOPK; ++it) {
    float lm = fmaxf(fmaxf(rem[0], rem[1]), fmaxf(rem[2], rem[3]));
    float gm = wmax32(lm);
    thresh = gm;
    unsigned long long ball = __ballot(lm == gm);
    int src = (int)__builtin_ctzll(ball);
    if (lane == src) {
      bool done = false;
#pragma unroll
      for (int i = 0; i < 4; ++i)
        if (!done && rem[i] == gm) { rem[i] = kNEG; done = true; }
    }
  }
  float mx = kNEG;
  float kept[4];
#pragma unroll
  for (int i = 0; i < 4; ++i) {
    kept[i] = (sc[i] >= thresh) ? sc[i] : kNEG;
    mx = fmaxf(mx, kept[i]);
  }
  mx = wmax32(mx);
  float sum = 0.0f, p[4];
#pragma unroll
  for (int i = 0; i < 4; ++i) { p[i] = __expf(kept[i] - mx); sum += p[i]; }
  sum = wsum32(sum);
  float rinv = 1.0f / sum;
#pragma unroll
  for (int i = 0; i < 4; ++i) pshare[lane + 32 * i] = p[i] * rinv;
  __syncthreads();
  float o0 = 0.0f, o1 = 0.0f;
  for (int sidx = 0; sidx < kWIN; ++sidx) {
    int key = kstart + sidx;
    if (key > q) break;
    float pv = pshare[sidx];
    const float* vrow = Vh + (size_t)key * (kHKV * kD);
    o0 += pv * vrow[lane];
    o1 += pv * vrow[lane + 32];
  }
  outk[(size_t)q * kHID + h * kD + lane] = o0;
  outk[(size_t)q * kHID + h * kD + lane + 32] = o1;
}

// ---------------- gate + blend ----------------
__global__ __launch_bounds__(32)
void gate_kernel(const float* __restrict__ pv, const float* __restrict__ pk,
                 const float* __restrict__ wg, const float* __restrict__ bg,
                 float* __restrict__ out) {
  int t = blockIdx.x, lane = threadIdx.x;
  const float* a = pv + (size_t)t * kHID;
  const float* b = pk + (size_t)t * kHID;
  float l0 = 0.0f, l1 = 0.0f;
  for (int i = lane; i < kHID; i += 32) {
    float av = a[i], bv = b[i];
    l0 += av * wg[i]            + bv * wg[kHID + i];
    l1 += av * wg[2 * kHID + i] + bv * wg[3 * kHID + i];
  }
  l0 = wsum32(l0) + bg[0];
  l1 = wsum32(l1) + bg[1];
  float w0 = 1.0f / (1.0f + __expf(l1 - l0));
  float w1 = 1.0f - w0;
  for (int i = lane; i < kHID; i += 32)
    out[(size_t)t * kHID + i] = w0 * a[i] + w1 * b[i];
}

static inline void cvt(const float* s, _Float16* d, int n, hipStream_t st) {
  int threads = 256;
  int blocks = (n / 4 + threads - 1) / threads;
  f32_to_f16<<<blocks, threads, 0, st>>>(s, d, n);
}

extern "C" void kernel_launch(void* const* d_in, const int* in_sizes, int n_in,
                              void* d_out, int out_size, void* d_ws, size_t ws_size,
                              hipStream_t stream) {
  const float* hidden = (const float*)d_in[0];
  const float* cosv   = (const float*)d_in[1];
  const float* sinv   = (const float*)d_in[2];
  const float* wq     = (const float*)d_in[3];
  const float* bq     = (const float*)d_in[4];
  const float* wk     = (const float*)d_in[5];
  const float* bk     = (const float*)d_in[6];
  const float* wv     = (const float*)d_in[7];
  const float* bv     = (const float*)d_in[8];
  const float* wo     = (const float*)d_in[9];
  const float* wq_k   = (const float*)d_in[10];
  const float* bq_k   = (const float*)d_in[11];
  const float* wk_k   = (const float*)d_in[12];
  const float* bk_k   = (const float*)d_in[13];
  const float* wv_k   = (const float*)d_in[14];
  const float* bv_k   = (const float*)d_in[15];
  const float* wo_k   = (const float*)d_in[16];
  const float* wg     = (const float*)d_in[17];
  const float* bg     = (const float*)d_in[18];
  const float* lsig   = (const float*)d_in[19];

  float* out  = (float*)d_out;
  float* attn = out + (size_t)kT * kHID;    // outputs concatenated: output, attn_v

  const size_t QSZ = (size_t)kT * kHID;     // 2,097,152
  const size_t KSZ = (size_t)kT * kHKV * kD;
  float* ws   = (float*)d_ws;
  // f32 scratch
  float* qb   = ws;
  float* kb   = qb + QSZ;
  float* vb   = kb + KSZ;
  float* qkb  = vb + KSZ;
  float* kkb  = qkb + QSZ;
  float* vkb  = kkb + KSZ;
  float* outv = vkb + KSZ;
  float* outk = outv + QSZ;
  float* pvb  = outk + QSZ;
  float* pkb  = pvb + QSZ;
  float* rm   = pkb + QSZ;
  float* rl   = rm + (size_t)kH * kT;
  // f16 scratch
  _Float16* h16   = (_Float16*)(rl + (size_t)kH * kT);
  _Float16* wq16  = h16 + QSZ;
  _Float16* wk16  = wq16 + QSZ;
  _Float16* wv16  = wk16 + KSZ;      // weight wk is (512,1024) = KSZ/2 elems
  _Float16* wqk16 = wv16 + KSZ;
  _Float16* wkk16 = wqk16 + QSZ;
  _Float16* wvk16 = wkk16 + KSZ;
  _Float16* wo16  = wvk16 + KSZ;
  _Float16* wok16 = wo16 + QSZ;
  _Float16* q16   = wok16 + QSZ;
  _Float16* k16   = q16 + QSZ;
  _Float16* v16   = k16 + KSZ;
  _Float16* ov16  = v16 + KSZ;
  _Float16* ok16  = ov16 + QSZ;

  // convert inputs/weights to f16 once
  cvt(hidden, h16, (int)QSZ, stream);
  cvt(wq,   wq16,  (int)QSZ, stream);
  cvt(wk,   wk16,  (int)KSZ, stream);
  cvt(wv,   wv16,  (int)KSZ, stream);
  cvt(wq_k, wqk16, (int)QSZ, stream);
  cvt(wk_k, wkk16, (int)KSZ, stream);
  cvt(wv_k, wvk16, (int)KSZ, stream);
  cvt(wo,   wo16,  (int)QSZ, stream);
  cvt(wo_k, wok16, (int)QSZ, stream);

  // QKV projections (6 WMMA GEMMs, 16x64 tiles)
  gemm_xwT16<<<dim3(kT / 16, kHID / 64), 32, 0, stream>>>(h16, wq16, bq, qb, kHID, kHID);
  gemm_xwT16<<<dim3(kT / 16, (kHKV * kD) / 64), 32, 0, stream>>>(h16, wk16, bk, kb, kHKV * kD, kHID);
  gemm_xwT16<<<dim3(kT / 16, (kHKV * kD) / 64), 32, 0, stream>>>(h16, wv16, bv, vb, kHKV * kD, kHID);
  gemm_xwT16<<<dim3(kT / 16, kHID / 64), 32, 0, stream>>>(h16, wqk16, bq_k, qkb, kHID, kHID);
  gemm_xwT16<<<dim3(kT / 16, (kHKV * kD) / 64), 32, 0, stream>>>(h16, wkk16, bk_k, kkb, kHKV * kD, kHID);
  gemm_xwT16<<<dim3(kT / 16, (kHKV * kD) / 64), 32, 0, stream>>>(h16, wvk16, bv_k, vkb, kHKV * kD, kHID);

  // RoPE (f32), then convert v-branch operands to f16 for attention WMMA
  rope_kernel<<<(kT * kH * 32 + 255) / 256, 256, 0, stream>>>(qb, cosv, sinv, kH);
  rope_kernel<<<(kT * kHKV * 32 + 255) / 256, 256, 0, stream>>>(kb, cosv, sinv, kHKV);
  rope_kernel<<<(kT * kH * 32 + 255) / 256, 256, 0, stream>>>(qkb, cosv, sinv, kH);
  rope_kernel<<<(kT * kHKV * 32 + 255) / 256, 256, 0, stream>>>(kkb, cosv, sinv, kHKV);
  cvt(qb, q16, (int)QSZ, stream);
  cvt(kb, k16, (int)KSZ, stream);
  cvt(vb, v16, (int)KSZ, stream);

  // v-branch attention: stats pass, then normalize + attn write + P@V
  attnv_stats<<<dim3(kT / 16, kH), 32, 0, stream>>>(q16, k16, rm, rl);
  attnv_out<<<dim3(kT / 16, kH), 32, 0, stream>>>(q16, k16, v16, rm, rl, attn, outv);

  // k-branch: windowed distance attention with top-32 threshold (f32, tiny)
  attnk_kernel<<<dim3(kT, kH), 32, 0, stream>>>(qkb, kkb, vkb, lsig, outk);

  // output projections (WMMA GEMMs) on f16 copies
  cvt(outv, ov16, (int)QSZ, stream);
  cvt(outk, ok16, (int)QSZ, stream);
  gemm_xwT16<<<dim3(kT / 16, kHID / 64), 32, 0, stream>>>(ov16, wo16, nullptr, pvb, kHID, kHID);
  gemm_xwT16<<<dim3(kT / 16, kHID / 64), 32, 0, stream>>>(ok16, wok16, nullptr, pkb, kHID, kHID);

  // gate + blend -> final output
  gate_kernel<<<kT, 32, 0, stream>>>(pvb, pkb, wg, bg, out);
}
